// HybridMemory_8186207666549
// MI455X (gfx1250) — compile-verified, matrix operand order checked by reading, not run on hardware
//
#include <hip/hip_runtime.h>
#include <hip/hip_bf16.h>

typedef __attribute__((ext_vector_type(16))) _Float16 v16h;
typedef __attribute__((ext_vector_type(8)))  _Float16 v8h;
typedef __attribute__((ext_vector_type(8)))  float    v8f;
typedef __attribute__((ext_vector_type(4)))  float    v4f;
typedef __attribute__((ext_vector_type(4)))  int      v4i;

#define BATCH 256
#define FEAT  2048
#define NMEM  65536
#define NCLUS 8000
#define INV_TEMP 20.0f

#define AS1 __attribute__((address_space(1)))
#define AS3 __attribute__((address_space(3)))

// Async global->LDS copy (ASYNCcnt path). Guarded: falls back to a synchronous
// register-staged LDS copy if the toolchain lacks the gfx1250 builtins.
// Probe round 2 revealed the builtin's parameter types: (v4i AS1*, v4i AS3*, Ii, Ii).
#if __has_builtin(__builtin_amdgcn_global_load_async_to_lds_b128) && \
    __has_builtin(__builtin_amdgcn_s_wait_asynccnt)
#define HAVE_ASYNC_LDS 1
#define ASYNC_COPY16(g, l)                                                    \
    __builtin_amdgcn_global_load_async_to_lds_b128(                           \
        (AS1 v4i*)(g), (AS3 v4i*)(l), 0, 0)
#define ASYNC_WAIT(n) __builtin_amdgcn_s_wait_asynccnt(n)
#else
#define HAVE_ASYNC_LDS 0
#define ASYNC_COPY16(g, l) (*(v4f*)(l) = *(const v4f*)(g))
#define ASYNC_WAIT(n) ((void)0)
#endif

// ---------------------------------------------------------------------------
// Workspace layout (bytes):
//   [0)        x_h  : 256*2048 f16 = 1,048,576
//   [1048576)  sim  : 8000*256 f32 = 8,192,000   (sim[c][b])
//   [9240576)  nums : 8000 u32     =    32,000
// ---------------------------------------------------------------------------
#define WS_XH   0
#define WS_SIM  1048576
#define WS_NUMS (1048576 + 8192000)

// ---- init: zero sim, nums, out --------------------------------------------
__global__ void hm_init(float* __restrict__ sim, unsigned* __restrict__ nums,
                        float* __restrict__ out) {
    unsigned i = blockIdx.x * 256u + threadIdx.x;   // 8000 blocks * 256 threads
    sim[i] = 0.0f;
    if (i < NCLUS) nums[i] = 0u;
    if (i == 0)    out[0]  = 0.0f;
}

// ---- normalize batch rows, fold 1/TEMP, emit f16 ---------------------------
__global__ __launch_bounds__(256) void hm_norm(const float* __restrict__ results,
                                               _Float16* __restrict__ xh) {
    __shared__ float red[256];
    const int b = blockIdx.x;
    const float* r = results + (size_t)b * FEAT;
    float v[8];
    float ss = 0.0f;
#pragma unroll
    for (int j = 0; j < 8; ++j) {
        v[j] = r[threadIdx.x + 256 * j];
        ss += v[j] * v[j];
    }
    red[threadIdx.x] = ss;
    __syncthreads();
    for (int s = 128; s > 0; s >>= 1) {
        if (threadIdx.x < s) red[threadIdx.x] += red[threadIdx.x + s];
        __syncthreads();
    }
    const float scale = rsqrtf(red[0]) * INV_TEMP;   // fold 1/TEMP into x
    _Float16* xo = xh + (size_t)b * FEAT;
#pragma unroll
    for (int j = 0; j < 8; ++j)
        xo[threadIdx.x + 256 * j] = (_Float16)(v[j] * scale);
}

// ---- cluster population counts --------------------------------------------
__global__ void hm_count(const int* __restrict__ labels, unsigned* __restrict__ nums) {
    int i = blockIdx.x * 256 + threadIdx.x;
    if (i < NMEM) atomicAdd(&nums[labels[i]], 1u);
}

// ---- WMMA operand loaders --------------------------------------------------
// A fragment: f16 row-major; lane holds M=l16, K chunks {k+8h..+7, k+16+8h..+7}
__device__ __forceinline__ v16h load_a_frag(const _Float16* __restrict__ rowp,
                                            int k, int half) {
    const _Float16* p = rowp + k + half * 8;
    v8h lo = *(const v8h*)(p);
    v8h hi = *(const v8h*)(p + 16);
    v16h a;
#pragma unroll
    for (int i = 0; i < 8; ++i) { a[i] = lo[i]; a[i + 8] = hi[i]; }
    return a;
}

// B fragment read from an LDS-staged fp32 row (32 floats), convert to f16.
__device__ __forceinline__ v16h load_b_frag_lds(const float* rowp, int half) {
    const float* p = rowp + half * 8;
    v4f c0 = *(const v4f*)(p);
    v4f c1 = *(const v4f*)(p + 4);
    v4f c2 = *(const v4f*)(p + 16);
    v4f c3 = *(const v4f*)(p + 20);
    v16h b;
#pragma unroll
    for (int i = 0; i < 4; ++i) {
        b[i]      = (_Float16)c0[i];
        b[i + 4]  = (_Float16)c1[i];
        b[i + 8]  = (_Float16)c2[i];
        b[i + 12] = (_Float16)c3[i];
    }
    return b;
}

// ---- fused GEMM (x_h · Fᵀ, f16 WMMA, fp32 acc) + label scatter -------------
// Block: M=256 (all batch) x N=64 memory rows; 8 waves, each owns a 64x32 tile
// (4x2 WMMA accumulators). Features are streamed HBM -> LDS with async b128
// copies, double-buffered per wave (4 KB per buffer), so a full K-step tile is
// in flight with zero VGPR cost; WMMA consumes the previous buffer from LDS.
__global__ __launch_bounds__(256) void hm_gemm_scatter(
        const _Float16* __restrict__ xh, const float* __restrict__ feats,
        const int* __restrict__ labels, float* __restrict__ sim) {
    __shared__ float lbs[2][8][1024];   // 2 bufs x 8 waves x (32 rows * 32 f32)

    const int lane = threadIdx.x & 31;
    const int w    = threadIdx.x >> 5;
    const int wm   = w & 3;          // M base = 64*wm
    const int wn   = w >> 2;         // N base = 32*wn
    const int half = lane >> 4;
    const int l16  = lane & 15;
    const int nBlock = blockIdx.x * 64;

    v8f acc[4][2];
#pragma unroll
    for (int tm = 0; tm < 4; ++tm)
#pragma unroll
        for (int tn = 0; tn < 2; ++tn) acc[tm][tn] = {};

    const _Float16* arow[4];
#pragma unroll
    for (int tm = 0; tm < 4; ++tm)
        arow[tm] = xh + (size_t)(64 * wm + 16 * tm + l16) * FEAT;
    const float* brow[2];
#pragma unroll
    for (int tn = 0; tn < 2; ++tn)
        brow[tn] = feats + (size_t)(nBlock + 32 * wn + 16 * tn + l16) * FEAT;

    // Issue one K-step tile: 8 async b128 copies (32 lanes x 16 B each).
    // Lane L copies row (L&15) of tile tn, 16-B chunk (L>>4)+2*i.
    auto issue_tile = [&](int k, int buf) {
        float* wbuf = &lbs[buf][w][0];
#pragma unroll
        for (int tn = 0; tn < 2; ++tn) {
#pragma unroll
            for (int i = 0; i < 4; ++i) {
                const int chunk = half + 2 * i;               // 0..7
                const float* g = brow[tn] + k + chunk * 4;
                float*       l = wbuf + (tn * 16 + l16) * 32 + chunk * 4;
                ASYNC_COPY16(g, l);
            }
        }
    };

    int buf = 0;
    issue_tile(0, 0);
    for (int k = 0; k < FEAT; k += 32) {
        if (k + 32 < FEAT) {
            issue_tile(k + 32, buf ^ 1);   // 8 more in flight
            ASYNC_WAIT(8);                 // current buffer's 8 are complete
            if (k + 160 <= FEAT) {         // pull further-ahead lines toward L2
                __builtin_prefetch(brow[0] + k + 128, 0, 1);
                __builtin_prefetch(brow[1] + k + 128, 0, 1);
            }
        } else {
            ASYNC_WAIT(0);
        }
        const float* wbuf = &lbs[buf][w][0];
        v16h b0 = load_b_frag_lds(wbuf + (0 * 16 + l16) * 32, half);
        v16h b1 = load_b_frag_lds(wbuf + (1 * 16 + l16) * 32, half);
#pragma unroll
        for (int tm = 0; tm < 4; ++tm) {
            v16h a = load_a_frag(arow[tm], k, half);
            acc[tm][0] = __builtin_amdgcn_wmma_f32_16x16x32_f16(
                false, a, false, b0, (short)0, acc[tm][0], false, false);
            acc[tm][1] = __builtin_amdgcn_wmma_f32_16x16x32_f16(
                false, a, false, b1, (short)0, acc[tm][1], false, false);
        }
        buf ^= 1;
    }

    // Scatter: D VGPR v -> (b = 64*wm + 16*tm + v + 8*half, m = nBlock+32*wn+16*tn+l16)
#pragma unroll
    for (int tn = 0; tn < 2; ++tn) {
        const int m = nBlock + 32 * wn + 16 * tn + l16;
        const int c = labels[m];
        float* simc = sim + (size_t)c * BATCH;
#pragma unroll
        for (int tm = 0; tm < 4; ++tm) {
            const int bbase = 64 * wm + 16 * tm + 8 * half;
#pragma unroll
            for (int v = 0; v < 8; ++v)
                atomicAdd(simc + bbase + v, acc[tm][tn][v]);
        }
    }
}

// ---- per-row masked softmax + NLL, mean over batch -------------------------
__global__ __launch_bounds__(256) void hm_loss(
        const float* __restrict__ sim, const unsigned* __restrict__ nums,
        const int* __restrict__ labels, const int* __restrict__ indexes,
        float* __restrict__ out) {
    __shared__ float red[256];
    __shared__ float etarget;
    __shared__ int   target_s;
    const int b = blockIdx.x;
    if (threadIdx.x == 0) {
        target_s = labels[indexes[b]];
        etarget  = 0.0f;
    }
    __syncthreads();
    const int target = target_s;
    float sum = 0.0f, et = 0.0f;
    for (int c = threadIdx.x; c < NCLUS; c += 256) {
        unsigned n = nums[c];
        if (n) {
            float s = sim[(size_t)c * BATCH + b] / (float)n;  // per-cluster mean
            float e = expf(s);                                 // reference: no max-sub
            sum += e;
            if (c == target) et = e;
        }
    }
    red[threadIdx.x] = sum;
    if (et != 0.0f) etarget = et;      // at most one writer
    __syncthreads();
    for (int s = 128; s > 0; s >>= 1) {
        if (threadIdx.x < s) red[threadIdx.x] += red[threadIdx.x + s];
        __syncthreads();
    }
    if (threadIdx.x == 0) {
        float prob = etarget / (red[0] + 1e-6f);
        float loss = -logf(prob + 1e-6f);
        atomicAdd(out, loss * (1.0f / (float)BATCH));
    }
}

// ---------------------------------------------------------------------------
extern "C" void kernel_launch(void* const* d_in, const int* in_sizes, int n_in,
                              void* d_out, int out_size, void* d_ws, size_t ws_size,
                              hipStream_t stream) {
    const float* results  = (const float*)d_in[0];   // [256, 2048] f32
    const int*   indexes  = (const int*)  d_in[1];   // [256] i32
    const float* features = (const float*)d_in[2];   // [65536, 2048] f32 (rows L2-normed)
    const int*   labels   = (const int*)  d_in[3];   // [65536] i32
    float* out = (float*)d_out;

    char* ws = (char*)d_ws;
    _Float16* xh   = (_Float16*)(ws + WS_XH);
    float*    sim  = (float*)   (ws + WS_SIM);
    unsigned* nums = (unsigned*)(ws + WS_NUMS);

    hm_init<<<NCLUS, 256, 0, stream>>>(sim, nums, out);
    hm_norm<<<BATCH, 256, 0, stream>>>(results, xh);
    hm_count<<<NMEM / 256, 256, 0, stream>>>(labels, nums);
    hm_gemm_scatter<<<NMEM / 64, 256, 0, stream>>>(xh, features, labels, sim);
    hm_loss<<<BATCH, 256, 0, stream>>>(sim, nums, labels, indexes, out);
}
